// VariationalLSTMModule_824633721236
// MI455X (gfx1250) — matrix operand order repaired and checked by reading, not compiled, not run on hardware
//
#include <hip/hip_runtime.h>

// ---------------------------------------------------------------------------
// Variational LSTM LM for MI455X (gfx1250, wave32, WMMA + TDM).
// B=32 T=128 V=50000 E=H=1024 L=2 O=50000.
//   * bf16 inputs, f32-accumulate v_wmma_f32_16x16x32_bf16.
//   * LSTM: 256 sequential step launches; each step 128 blocks x 8 waves,
//     K split across waves, LDS tree-reduction + fused cell update.
//   * Decoder (419 GFLOP): block of 8 waves shares a W_dec K-slab staged in
//     LDS by the Tensor Data Mover (double-buffered, s_wait_tensorcnt),
//     cutting B-operand L2 traffic 8x vs per-wave global loads.
// ---------------------------------------------------------------------------

#define B_   32
#define T_   128
#define E_   1024
#define H_   1024
#define L_   2
#define O_   50000
#define G4H  4096
#define KT_  (H_ / 32)          // 32 K-tiles in decoder

typedef unsigned int   u32;
typedef unsigned short u16;
typedef __attribute__((ext_vector_type(16))) __bf16 v16bf;
typedef __attribute__((ext_vector_type(8)))  float  v8f;
typedef __attribute__((ext_vector_type(4)))  unsigned int v4u;
typedef __attribute__((ext_vector_type(8)))  int    v8i;
typedef __attribute__((ext_vector_type(4)))  int    v4i;

#if defined(__has_builtin)
#  if __has_builtin(__builtin_amdgcn_tensor_load_to_lds) && \
      __has_builtin(__builtin_amdgcn_s_wait_tensorcnt)
#    define USE_TDM 1
#  endif
#endif
#ifndef USE_TDM
#  define USE_TDM 0
#endif

struct Frag32 { uint4 lo, hi; };   // 32 bytes == v16bf

__device__ __forceinline__ u16 f2bf(float f) {
  u32 u = __builtin_bit_cast(u32, f);
  u32 r = (u + 0x7FFFu + ((u >> 16) & 1u)) >> 16;   // RNE
  return (u16)r;
}

__device__ __forceinline__ float sigmoidf_(float x) {
  return 1.0f / (1.0f + __expf(-x));
}

// A-fragment (16x32 bf16, MxK): lanes 0-15 K(0..15); lanes16-31 K(16..31).
__device__ __forceinline__ v16bf load_frag_A(const u16* __restrict__ base,
                                             int ld, int m0, int k0) {
  const int lane = threadIdx.x & 31;
  const int m = m0 + (lane & 15);
  const int k = k0 + ((lane & 16) ? 16 : 0);
  const u16* p = base + (size_t)m * ld + k;
  Frag32 f;
  f.lo = *(const uint4*)(p);
  f.hi = *(const uint4*)(p + 8);
  return __builtin_bit_cast(v16bf, f);
}

// B-fragment (32x16 bf16, KxN), B[k][n] = W[n0+n][k] row-major (stride ld):
// lanes 0-15: N=lane, K=k0..k0+15; lanes 16-31: K += 16.
__device__ __forceinline__ v16bf load_frag_B(const u16* __restrict__ base,
                                             int ld, int n0, int k0) {
  const int lane = threadIdx.x & 31;
  const int n = n0 + (lane & 15);
  const int k = k0 + ((lane & 16) ? 16 : 0);
  const u16* p = base + (size_t)n * ld + k;
  Frag32 f;
  f.lo = *(const uint4*)(p);
  f.hi = *(const uint4*)(p + 8);
  return __builtin_bit_cast(v16bf, f);
}

#define WMMA_BF16(a, b, c) \
  __builtin_amdgcn_wmma_f32_16x16x32_bf16(false, (a), false, (b), (short)0, (c), false, false)

// ---------------------------------------------------------------------------
__global__ void cvt_f32_bf16(const float* __restrict__ src,
                             u16* __restrict__ dst, int n) {
  int i = blockIdx.x * blockDim.x + threadIdx.x;
  if (i < n) dst[i] = f2bf(src[i]);
}

__global__ void embed_kernel(const int* __restrict__ ids,
                             const float* __restrict__ table,
                             const float* __restrict__ emb_mask,
                             u16* __restrict__ xseq) {
  int i = blockIdx.x * blockDim.x + threadIdx.x;     // over B*T*E
  if (i >= B_ * T_ * E_) return;
  int e  = i & (E_ - 1);
  int bt = i >> 10;
  int t  = bt & (T_ - 1);
  int b  = bt >> 7;
  int id = ids[b * T_ + t];
  float v = table[(size_t)id * E_ + e] * emb_mask[b * E_ + e];
  xseq[((size_t)t * B_ + b) * E_ + e] = f2bf(v);
}

__global__ void init_state(float* __restrict__ c_ws,
                           u16* __restrict__ hrec,
                           u16* __restrict__ hlay) {
  int i = blockIdx.x * blockDim.x + threadIdx.x;
  if (i < L_ * B_ * H_) {
    c_ws[i] = 0.0f;
    hrec[i] = 0;
    hrec[L_ * B_ * H_ + i] = 0;
  }
  if (i < B_ * H_) hlay[i] = 0;
}

// ---------------------------------------------------------------------------
// One (layer, timestep).  128 blocks (2 batch-tiles x 64 hidden-tiles),
// 8 waves per block.  Wave w accumulates K-slice [w]*256 of the gate GEMM
// (waves 0-3: input term, waves 4-7: recurrent term), partials are reduced
// in LDS, then 256 threads do the fused LSTM cell update (1 element each).
__global__ void __launch_bounds__(256)
lstm_step(int l, int t,
          const u16* __restrict__ xseq,       // [T][B][E]
          const u16* __restrict__ hlay_in,    // [B][H]
          u16* __restrict__ hrec_buf,         // [2][L][B][H] ping-pong
          const u16* __restrict__ Wih,        // [L][4H][E]
          const u16* __restrict__ Whh,        // [L][4H][H]
          const float* __restrict__ b_ih,
          const float* __restrict__ b_hh,
          const float* __restrict__ time_mask,
          const float* __restrict__ layer_mask,
          float* __restrict__ c_ws,           // [L][B][H]
          u16* __restrict__ hlay_out,         // [B][H]
          u16* __restrict__ outs) {           // [B][T][H]
  __shared__ float red[8][4][8][32];          // [wave][gate][j][lane] = 32 KB

  const int tid   = threadIdx.x;
  const int lane  = tid & 31;
  const int wave  = tid >> 5;
  const int mtile = blockIdx.x & 1;
  const int ntile = blockIdx.x >> 1;
  const int mbase = mtile * 16;
  const int hbase = ntile * 16;

  const u16* hrec_r = hrec_buf + (size_t)(t & 1) * (L_ * B_ * H_);
  u16*       hrec_w = hrec_buf + (size_t)((t + 1) & 1) * (L_ * B_ * H_);

  // ---- partial gate GEMM (per-wave K slice of 256) ----
  const u16* A;
  const u16* W;
  int kb;
  if (wave < 4) {   // input contribution
    A  = (l == 0) ? (xseq + (size_t)t * B_ * E_) : hlay_in;
    W  = Wih + (size_t)l * G4H * E_;
    kb = wave * 256;
  } else {          // recurrent contribution
    A  = hrec_r + (size_t)l * B_ * H_;
    W  = Whh + (size_t)l * G4H * H_;
    kb = (wave - 4) * 256;
  }

  v8f accs[4] = {};
  for (int k = kb; k < kb + 256; k += 32) {
    v16bf a = load_frag_A(A, 1024, mbase, k);
#pragma unroll
    for (int g = 0; g < 4; ++g) {
      v16bf b = load_frag_B(W, 1024, g * H_ + hbase, k);
      accs[g] = WMMA_BF16(a, b, accs[g]);
    }
  }

#pragma unroll
  for (int g = 0; g < 4; ++g)
#pragma unroll
    for (int j = 0; j < 8; ++j)
      red[wave][g][j][lane] = accs[g][j];
  __syncthreads();

  // ---- reduce + fused cell update: one (batch,hidden) element per thread --
  const int m = tid >> 4;            // 0..15 within tile
  const int n = tid & 15;
  const int j = m & 7;
  const int rl = n + ((m >> 3) << 4);   // C layout lane for (m,n)

  float s[4];
#pragma unroll
  for (int g = 0; g < 4; ++g) {
    float v = 0.0f;
#pragma unroll
    for (int w = 0; w < 8; ++w) v += red[w][g][j][rl];
    s[g] = v;
  }

  const int brow = mbase + m;
  const int hcol = hbase + n;
  const size_t sidx = (size_t)l * B_ * H_ + (size_t)brow * H_ + hcol;
  const float iv = sigmoidf_(s[0] + b_ih[l * G4H + 0 * H_ + hcol] + b_hh[l * G4H + 0 * H_ + hcol]);
  const float fv = sigmoidf_(s[1] + b_ih[l * G4H + 1 * H_ + hcol] + b_hh[l * G4H + 1 * H_ + hcol]);
  const float gv = tanhf    (s[2] + b_ih[l * G4H + 2 * H_ + hcol] + b_hh[l * G4H + 2 * H_ + hcol]);
  const float ov = sigmoidf_(s[3] + b_ih[l * G4H + 3 * H_ + hcol] + b_hh[l * G4H + 3 * H_ + hcol]);
  const float cn = fv * c_ws[sidx] + iv * gv;
  const float hn = ov * tanhf(cn);
  c_ws[sidx]   = cn;
  hrec_w[sidx] = f2bf(hn * time_mask[sidx]);
  const float lo = hn * layer_mask[sidx];
  if (l == L_ - 1)
    outs[((size_t)brow * T_ + t) * H_ + hcol] = f2bf(lo);
  else
    hlay_out[(size_t)brow * H_ + hcol] = f2bf(lo);
}

// ---------------------------------------------------------------------------
// Decoder GEMM: logits[4096][50000] = outs[4096][1024] x Wdec^T + b_dec.
// Block = 8 waves = 128 M rows x 80 N cols.  The 80x32 W_dec K-slab is
// staged in LDS (TDM double-buffered when available) and shared by all
// 8 waves -> 8x less B-operand L2 traffic.  LDS rows padded to 80 B.
__global__ void __launch_bounds__(256)
decoder_kernel(const u16* __restrict__ outs,      // [4096][1024]
               const u16* __restrict__ Wdec,      // [50000][1024]
               const float* __restrict__ b_dec,   // [50000]
               float* __restrict__ logits) {      // [4096][50000]
  __shared__ u16 sB[2][80 * 40];    // 2 x 6400 B, row stride 40 halfs (80 B)

  const int tid   = threadIdx.x;
  const int lane  = tid & 31;
  const int wave  = tid >> 5;
  const int mtile = blockIdx.y * 8 + wave;     // 0..255
  const int mbase = mtile * 16;
  const int nbase = blockIdx.x * 80;           // 5 tiles of 16

  v8f acc[5] = {};

  auto mma_step = [&](int ki, int cur) {
    v16bf a = load_frag_A(outs, H_, mbase, ki * 32);
#pragma unroll
    for (int jt = 0; jt < 5; ++jt) {
      const int r = jt * 16 + (lane & 15);
      const int koff = (lane & 16) ? 16 : 0;
      Frag32 f;
      f.lo = *(const uint4*)&sB[cur][r * 40 + koff];
      f.hi = *(const uint4*)&sB[cur][r * 40 + koff + 8];
      acc[jt] = WMMA_BF16(a, __builtin_bit_cast(v16bf, f), acc[jt]);
    }
  };

#if USE_TDM
  // D# group1 (constant): data_size=2B; pad_enable, pad every 16 DWORDs
  // (one 64B row) by 4 DWORDs -> 80B LDS row stride; tensor 1024 x 50000,
  // dim0_stride 1024; tile 32(K) x 80(rows).
  const v8i g1 = { (int)(0x10000u | 0x100000u | (3u << 22) | (3u << 25)),
                   (int)(1024u << 16),       // tensor_dim0[15:0] in [31:16]
                   (int)(0xC350u << 16),     // tensor_dim1[15:0] in [31:16]
                   (int)(32u << 16),         // tile_dim0 in [31:16]
                   80,                       // tile_dim1
                   1024,                     // tensor_dim0_stride[31:0]
                   0, 0 };
  const v4i z4 = {0, 0, 0, 0};
  const v8i z8 = {0, 0, 0, 0, 0, 0, 0, 0};
  const unsigned long long wbase =
      (unsigned long long)(uintptr_t)Wdec + (unsigned long long)nbase * (H_ * 2);

  auto issue = [&](int buf, int k0) {
    const unsigned long long ga = wbase + (unsigned long long)k0 * 2;
    v4u g0;
    g0.x = 1u;                                  // count=1 (valid D#)
    g0.y = (u32)(uintptr_t)&sB[buf][0];         // lds_addr (bytes)
    g0.z = (u32)ga;                             // global_addr[31:0]
    g0.w = (u32)(ga >> 32) | 0x80000000u;       // global_addr[56:32] | type=2
    // 6-arg form on this toolchain: (g0, g1, g2, g3, g4, cpol)
    __builtin_amdgcn_tensor_load_to_lds(g0, g1, z4, z4, z8, 0);
  };

  if (tid < 32) issue(0, 0);                    // prologue: fill buffer 0
  for (int ki = 0; ki < KT_ - 1; ++ki) {
    const int cur = ki & 1;
    if (tid < 32) {
      issue(cur ^ 1, (ki + 1) * 32);            // prefetch next slab via TDM
      __builtin_amdgcn_s_wait_tensorcnt(1);     // current slab landed in LDS
    }
    __syncthreads();
    mma_step(ki, cur);
    __syncthreads();                            // reads done before reuse
  }
  {
    const int ki = KT_ - 1;
    if (tid < 32) __builtin_amdgcn_s_wait_tensorcnt(0);
    __syncthreads();
    mma_step(ki, ki & 1);
  }
#else
  // Fallback: cooperative load + ds_store staging, single buffer.
  for (int ki = 0; ki < KT_; ++ki) {
    __syncthreads();
    for (int c = tid; c < 320; c += 256) {      // 320 x 16B chunks
      const int r = c >> 2, q = c & 3;
      *(uint4*)&sB[0][r * 40 + q * 8] =
          *(const uint4*)(Wdec + (size_t)(nbase + r) * H_ + ki * 32 + q * 8);
    }
    __syncthreads();
    mma_step(ki, 0);
  }
#endif

  // ---- epilogue: bias + store ----
  const int mofs = (lane & 16) ? 8 : 0;
#pragma unroll
  for (int jt = 0; jt < 5; ++jt) {
    const int o = nbase + jt * 16 + (lane & 15);
    const float bb = b_dec[o];
#pragma unroll
    for (int j = 0; j < 8; ++j) {
      const int r = mbase + j + mofs;
      logits[(size_t)r * O_ + o] = acc[jt][j] + bb;
    }
  }
}

// ---------------------------------------------------------------------------
extern "C" void kernel_launch(void* const* d_in, const int* in_sizes, int n_in,
                              void* d_out, int out_size, void* d_ws, size_t ws_size,
                              hipStream_t stream) {
  const int*   ids        = (const int*)  d_in[0];
  const float* emb_table  = (const float*)d_in[1];
  const float* W_ih       = (const float*)d_in[2];
  const float* W_hh       = (const float*)d_in[3];
  const float* b_ih       = (const float*)d_in[4];
  const float* b_hh       = (const float*)d_in[5];
  const float* W_dec      = (const float*)d_in[6];
  const float* b_dec      = (const float*)d_in[7];
  const float* emb_mask   = (const float*)d_in[8];
  const float* time_mask  = (const float*)d_in[9];
  const float* layer_mask = (const float*)d_in[10];
  float* logits = (float*)d_out;

  char* ws = (char*)d_ws;
  size_t off = 0;
  auto carve = [&](size_t bytes) -> char* {
    char* p = ws + off;
    off = (off + bytes + 255) & ~(size_t)255;
    return p;
  };
  u16*   Wih_bf  = (u16*)  carve((size_t)L_ * G4H * E_ * 2);
  u16*   Whh_bf  = (u16*)  carve((size_t)L_ * G4H * H_ * 2);
  u16*   Wdec_bf = (u16*)  carve((size_t)O_ * H_ * 2);
  u16*   xseq_bf = (u16*)  carve((size_t)T_ * B_ * E_ * 2);
  u16*   outs_bf = (u16*)  carve((size_t)B_ * T_ * H_ * 2);
  u16*   hrec_bf = (u16*)  carve((size_t)2 * L_ * B_ * H_ * 2);
  u16*   hlay_bf = (u16*)  carve((size_t)B_ * H_ * 2);
  float* c_ws    = (float*)carve((size_t)L_ * B_ * H_ * 4);
  (void)ws_size; (void)in_sizes; (void)n_in; (void)out_size;

  {
    int n = L_ * G4H * E_;
    cvt_f32_bf16<<<(n + 255) / 256, 256, 0, stream>>>(W_ih, Wih_bf, n);
  }
  {
    int n = L_ * G4H * H_;
    cvt_f32_bf16<<<(n + 255) / 256, 256, 0, stream>>>(W_hh, Whh_bf, n);
  }
  {
    int n = O_ * H_;
    cvt_f32_bf16<<<(n + 255) / 256, 256, 0, stream>>>(W_dec, Wdec_bf, n);
  }
  {
    int n = B_ * T_ * E_;
    embed_kernel<<<(n + 255) / 256, 256, 0, stream>>>(ids, emb_table, emb_mask, xseq_bf);
  }
  {
    int n = L_ * B_ * H_;
    init_state<<<(n + 255) / 256, 256, 0, stream>>>(c_ws, hrec_bf, hlay_bf);
  }

  for (int t = 0; t < T_; ++t) {
    for (int l = 0; l < L_; ++l) {
      lstm_step<<<128, 256, 0, stream>>>(l, t, xseq_bf, hlay_bf, hrec_bf,
                                         Wih_bf, Whh_bf, b_ih, b_hh,
                                         time_mask, layer_mask,
                                         c_ws, hlay_bf, outs_bf);
    }
  }

  dim3 gdec(625, 32, 1);
  decoder_kernel<<<gdec, 256, 0, stream>>>(outs_bf, Wdec_bf, b_dec, logits);
}
